// Bonv_89369679495333
// MI455X (gfx1250) — matrix operand order repaired and checked by reading, not compiled, not run on hardware
//
#include <hip/hip_runtime.h>
#include <hip/hip_bf16.h>
#include <math.h>

#define NN 8192
#define CC 128
#define EPSILON 1e-15f

typedef __attribute__((ext_vector_type(16))) __bf16 v16bf;
typedef __attribute__((ext_vector_type(8)))  float  v8f;

union FragBF { uint4 u[2]; v16bf v; };

__device__ __forceinline__ unsigned short f2bf(float f) {
    unsigned int u = __float_as_uint(f);
    u += 0x7fffu + ((u >> 16) & 1u);           // round-to-nearest-even
    return (unsigned short)(u >> 16);
}

// ------------------------------------------------- pass 1 over A (cols)
// grid (16 col-groups, 4 row-chunks). Each thread owns 2 adjacent columns
// (float2 loads). Deterministic per-chunk partials, no atomics.
__global__ void k_colreduce(const float* __restrict__ A,
                            const float* __restrict__ nodes,
                            float* __restrict__ degP, float* __restrict__ aggP,
                            float* __restrict__ colPart) {
    __shared__ float nd[512];
    __shared__ float red[256];
    int t = threadIdx.x;
    int j = (blockIdx.x * 256 + t) * 2;
    int p = blockIdx.y;                         // row chunk: 2048 rows
    int ibeg = p * (NN/4), iend = ibeg + (NN/4);
    float dg0=0.f, dg1=0.f, ax0=0.f, ax1=0.f, ay0=0.f, ay1=0.f;
    for (int i0 = ibeg; i0 < iend; i0 += 256) {
        nd[2*t]   = nodes[2*(i0+t)];
        nd[2*t+1] = nodes[2*(i0+t)+1];
        __syncthreads();
        for (int r = 0; r < 256; ++r) {
            float2 a = *reinterpret_cast<const float2*>(&A[(long)(i0+r)*NN + j]);
            float nx = nd[2*r], ny = nd[2*r+1];
            dg0 += a.x; dg1 += a.y;
            ax0 += a.x*nx; ax1 += a.y*nx;
            ay0 += a.x*ny; ay1 += a.y*ny;
        }
        __syncthreads();
    }
    degP[p*NN + j] = dg0; degP[p*NN + j+1] = dg1;
    aggP[p*2*NN + 2*j]   = ax0; aggP[p*2*NN + 2*j+1] = ay0;
    aggP[p*2*NN + 2*j+2] = ax1; aggP[p*2*NN + 2*j+3] = ay1;
    red[t] = dg0 + dg1; __syncthreads();
    for (int s2 = 128; s2 > 0; s2 >>= 1) { if (t < s2) red[t] += red[t+s2]; __syncthreads(); }
    if (t == 0) colPart[p*16 + blockIdx.x] = red[0];
}

__global__ void k_degred(const float* __restrict__ degP, const float* __restrict__ aggP,
                         float* __restrict__ deg, float* __restrict__ aggT) {
    int j = blockIdx.x * 256 + threadIdx.x;     // 32 blocks
    float d = 0.f, ax = 0.f, ay = 0.f;
    for (int p = 0; p < 4; ++p) {
        d  += degP[p*NN + j];
        ax += aggP[p*2*NN + 2*j];
        ay += aggP[p*2*NN + 2*j+1];
    }
    deg[j] = d; aggT[2*j] = ax; aggT[2*j+1] = ay;
}

// --------------------------------- sage1 (x), sage2 logits + softmax (s)
// One wave per node. Writes s in f32 and pre-swizzled bf16 WMMA fragment
// layouts: sA = A-operand frags of s^T, sB = B-operand frags of s.
__global__ void k_softmax(const float* __restrict__ nodes,
                          const float* __restrict__ deg,
                          const float* __restrict__ aggT,
                          const float* __restrict__ w1l, const float* __restrict__ b1,
                          const float* __restrict__ w1r,
                          const float* __restrict__ w2l, const float* __restrict__ b2,
                          const float* __restrict__ w2r,
                          float* __restrict__ s_f32,
                          unsigned short* __restrict__ sA,
                          unsigned short* __restrict__ sB,
                          float* __restrict__ x, float* __restrict__ entPart) {
    __shared__ float entw[8];
    int t = threadIdx.x, lane = t & 31, wv = t >> 5;
    int i = blockIdx.x * 8 + wv;
    float inv = 1.0f / fmaxf(deg[i], 1.0f);
    float a0 = aggT[2*i] * inv, a1 = aggT[2*i+1] * inv;
    float n0 = nodes[2*i],      n1 = nodes[2*i+1];
    if (lane == 0) {
        x[2*i]   = a0*w1l[0] + a1*w1l[2] + b1[0] + n0*w1r[0] + n1*w1r[2];
        x[2*i+1] = a0*w1l[1] + a1*w1l[3] + b1[1] + n0*w1r[1] + n1*w1r[3];
    }
    float lg[4];
    float mx = -3.0e38f;
    for (int q = 0; q < 4; ++q) {
        int c = lane + 32*q;
        lg[q] = a0*w2l[c] + a1*w2l[CC+c] + b2[c] + n0*w2r[c] + n1*w2r[CC+c];
        mx = fmaxf(mx, lg[q]);
    }
    for (int m2 = 16; m2 > 0; m2 >>= 1) mx = fmaxf(mx, __shfl_xor(mx, m2, 32));
    float sum = 0.f;
    for (int q = 0; q < 4; ++q) { lg[q] = __expf(lg[q] - mx); sum += lg[q]; }
    for (int m2 = 16; m2 > 0; m2 >>= 1) sum += __shfl_xor(sum, m2, 32);
    float isum = 1.0f / sum;

    int kt = i >> 5, r = i & 31;                            // K-tile / row-in-tile
    int h  = r >> 4, eA = r & 15;                           // A-frag slot
    int laneB = 16 * (r >> 4), eB = r & 15;                 // B-frag slot
    float ent = 0.f;
    for (int q = 0; q < 4; ++q) {
        int c = lane + 32*q;
        float sv = lg[q] * isum;
        ent += -sv * __logf(sv + EPSILON);
        s_f32[(long)i*CC + c] = sv;
        unsigned short bv = f2bf(sv);
        int mt = c >> 4, m = c & 15;
        sA[(((long)(kt*8 + mt))*32 + (m + 16*h))*16 + eA] = bv;
        sB[(((long)(kt*8 + mt))*32 + (m + laneB))*16 + eB] = bv;
    }
    for (int m2 = 16; m2 > 0; m2 >>= 1) ent += __shfl_xor(ent, m2, 32);
    if (lane == 0) entw[wv] = ent;
    __syncthreads();
    if (t == 0) {
        float e = 0.f;
        for (int q = 0; q < 8; ++q) e += entw[q];
        entPart[blockIdx.x] = e;
    }
}

// --------------------------------------------- M = A @ s  (bf16 WMMA)
// Block = 16 rows x 128 cols (8 waves). K=64 per macro-iter: each thread
// stages one float4 (b128 load -> packed b64 ds_store), 2 WMMAs per barrier,
// double-buffered LDS, unconditional speculative prefetch.
// Epilogue fuses tAs partial = sum(s .* M) and writes M in B-frag layout.
__global__ void k_AxS(const float* __restrict__ A,
                      const unsigned short* __restrict__ sB,
                      const float* __restrict__ s_f32,
                      unsigned short* __restrict__ MB,
                      float* __restrict__ tAsPart) {
    __shared__ alignas(16) unsigned short ldsA[2][1024];
    __shared__ float red[256];
    int t = threadIdx.x, lane = t & 31, w = t >> 5;
    int rt = blockIdx.x, row0 = rt * 16;

    // staging: thread t owns k=kk..kk+3 of row m (16 threads x 4 floats = 64 K)
    int m  = t >> 4;
    int kk = (4*t) & 63;
    int k  = kk & 31;
    int hf = kk >> 5;                               // which 32-wide sub-tile
    int h  = k >> 4;
    int e  = k & 15;                                // multiple of 4
    int ldsOff = hf*512 + (m + 16*h)*16 + e;        // ushort index (x4 -> 8B aligned)
    const float* arow = A + (long)(row0 + m)*NN + kk;

    v8f acc = {0.f,0.f,0.f,0.f,0.f,0.f,0.f,0.f};
    for (int it = 0; it < NN/64; ++it) {
        float4 av = *reinterpret_cast<const float4*>(arow + it*64);
        __builtin_prefetch(arow + (it+4)*64, 0, 3);   // speculative: safe unconditionally
        uint2 pk2;
        pk2.x = (unsigned)f2bf(av.x) | ((unsigned)f2bf(av.y) << 16);
        pk2.y = (unsigned)f2bf(av.z) | ((unsigned)f2bf(av.w) << 16);
        *reinterpret_cast<uint2*>(&ldsA[it & 1][ldsOff]) = pk2;
        __syncthreads();
        for (int sub = 0; sub < 2; ++sub) {
            FragBF fa, fb;
            const uint4* pa = reinterpret_cast<const uint4*>(&ldsA[it & 1][sub*512 + lane*16]);
            fa.u[0] = pa[0]; fa.u[1] = pa[1];
            int kt = it*2 + sub;
            const uint4* pb = reinterpret_cast<const uint4*>(&sB[(((long)(kt*8 + w))*32 + lane)*16]);
            fb.u[0] = pb[0]; fb.u[1] = pb[1];
            acc = __builtin_amdgcn_wmma_f32_16x16x32_bf16(false, fa.v, false, fb.v,
                                                          (short)0, acc, false, false);
        }
    }
    // epilogue: tile is M[row0 + r + 8*(lane>>4)][w*16 + (lane&15)]
    int col = w*16 + (lane & 15);
    int half = lane >> 4;
    float tl = 0.f;
    unsigned short hb[8];
    for (int r = 0; r < 8; ++r) {
        int row = row0 + r + 8*half;
        tl += acc[r] * s_f32[(long)row*CC + col];
        hb[r] = f2bf(acc[r]);
    }
    int kt2 = rt >> 1;
    int laneDst = (lane & 15) + 16*(rt & 1);
    int e0 = 8*half;
    uint4 pk;
    pk.x = (unsigned)hb[0] | ((unsigned)hb[1] << 16);
    pk.y = (unsigned)hb[2] | ((unsigned)hb[3] << 16);
    pk.z = (unsigned)hb[4] | ((unsigned)hb[5] << 16);
    pk.w = (unsigned)hb[6] | ((unsigned)hb[7] << 16);
    *reinterpret_cast<uint4*>(&MB[(((long)(kt2*8 + w))*32 + laneDst)*16 + e0]) = pk;
    red[t] = tl; __syncthreads();
    for (int s2 = 128; s2 > 0; s2 >>= 1) { if (t < s2) red[t] += red[t+s2]; __syncthreads(); }
    if (t == 0) tAsPart[blockIdx.x] = red[0];
}

// ------------- Part[kc] += s^T-chunk @ B-chunk (split-K x8, deterministic)
__global__ void k_sTB(const unsigned short* __restrict__ Af,
                      const unsigned short* __restrict__ Bf,
                      float* __restrict__ Part) {
    int t = threadIdx.x, lane = t & 31, ct = t >> 5;
    int mt = blockIdx.x, kc = blockIdx.y;
    v8f acc = {0.f,0.f,0.f,0.f,0.f,0.f,0.f,0.f};
    for (int kt = kc*32; kt < kc*32 + 32; ++kt) {
        FragBF fa, fb;
        const uint4* pa = reinterpret_cast<const uint4*>(&Af[(((long)(kt*8 + mt))*32 + lane)*16]);
        fa.u[0] = pa[0]; fa.u[1] = pa[1];
        const uint4* pb = reinterpret_cast<const uint4*>(&Bf[(((long)(kt*8 + ct))*32 + lane)*16]);
        fb.u[0] = pb[0]; fb.u[1] = pb[1];
        acc = __builtin_amdgcn_wmma_f32_16x16x32_bf16(false, fa.v, false, fb.v,
                                                      (short)0, acc, false, false);
    }
    int col = ct*16 + (lane & 15);
    int half = lane >> 4;
    for (int r = 0; r < 8; ++r)
        Part[kc*CC*CC + (mt*16 + r + 8*half)*CC + col] = acc[r];
}

__global__ void k_red8(const float* __restrict__ part, float* __restrict__ outm) {
    int i = blockIdx.x * 256 + threadIdx.x;     // 64 blocks
    float v = 0.f;
    for (int p = 0; p < 8; ++p) v += part[p*CC*CC + i];
    outm[i] = v;
}

// ---------------------- x_out = s^T @ x, coalesced partials (no atomics)
__global__ void k_xout(const float* __restrict__ s_f32,
                       const float* __restrict__ x, float* __restrict__ xop) {
    __shared__ float rx[256], ry[256];
    int t = threadIdx.x;
    int c = t & 127, g = t >> 7;                 // g in {0,1}
    float px = 0.f, py = 0.f;
    for (int i = blockIdx.x*2 + g; i < NN; i += 32) {
        float sv = s_f32[(long)i*CC + c];
        px += sv * x[2*i]; py += sv * x[2*i+1];
    }
    rx[t] = px; ry[t] = py;
    __syncthreads();
    if (t < 128) {
        xop[blockIdx.x*256 + 2*c]     = rx[t] + rx[t+128];
        xop[blockIdx.x*256 + 2*c + 1] = ry[t] + ry[t+128];
    }
}

__global__ void k_xored(const float* __restrict__ xop, float* __restrict__ xo) {
    int t = threadIdx.x;                          // 256 threads
    float v = 0.f;
    for (int b = 0; b < 16; ++b) v += xop[b*256 + t];
    xo[t] = v;
}

// ----------- losses: ||A-ssT||^2 = sumA - 2*tAs + ||G||_F^2  (deterministic)
__global__ void k_scalars(const float* __restrict__ G,
                          const float* __restrict__ colPart,
                          const float* __restrict__ tAsPart,
                          const float* __restrict__ entPart,
                          float* __restrict__ out) {
    __shared__ float red[256];
    __shared__ float res[4];
    int t = threadIdx.x;

    float p = 0.f;                               // ||G||_F^2
    for (int i = t; i < CC*CC; i += 256) { float g = G[i]; p += g*g; }
    red[t] = p; __syncthreads();
    for (int s2 = 128; s2 > 0; s2 >>= 1) { if (t < s2) red[t] += red[t+s2]; __syncthreads(); }
    if (t == 0) res[0] = red[0];
    __syncthreads();

    red[t] = (t < 64) ? colPart[t] : 0.f;        // sumA
    __syncthreads();
    for (int s2 = 128; s2 > 0; s2 >>= 1) { if (t < s2) red[t] += red[t+s2]; __syncthreads(); }
    if (t == 0) res[1] = red[0];
    __syncthreads();

    red[t] = tAsPart[t] + tAsPart[t+256];        // tAs
    __syncthreads();
    for (int s2 = 128; s2 > 0; s2 >>= 1) { if (t < s2) red[t] += red[t+s2]; __syncthreads(); }
    if (t == 0) res[2] = red[0];
    __syncthreads();

    red[t] = entPart[t] + entPart[t+256] + entPart[t+512] + entPart[t+768];
    __syncthreads();
    for (int s2 = 128; s2 > 0; s2 >>= 1) { if (t < s2) red[t] += red[t+s2]; __syncthreads(); }
    if (t == 0) {
        float fro2 = res[1] - 2.0f*res[2] + res[0];
        out[384] = sqrtf(fmaxf(fro2, 0.0f)) / ((float)NN * (float)NN);
        out[385] = red[0] / (float)NN;
    }
}

// ------------- argmax one-hot E from adj_pool, edge_index, final SAGE
__global__ void k_final(const float* __restrict__ P, const float* __restrict__ xo,
                        const float* __restrict__ w3l, const float* __restrict__ b3,
                        const float* __restrict__ w3r, float* __restrict__ out) {
    __shared__ float rmax[128];
    int c = threadIdx.x;
    float m = -3.4e38f; int arg = 0;
    for (int j = 0; j < CC; ++j) {
        float v = P[c*CC + j];
        if (v > m) { m = v; arg = j; }
    }
    rmax[c] = m;
    out[128 + c] = (float)c;        // edge_index row 0
    out[256 + c] = (float)arg;      // edge_index row 1
    __syncthreads();
    float dg = 0.f, ax = 0.f, ay = 0.f;
    for (int r = 0; r < CC; ++r) {
        float e = (P[r*CC + c] == rmax[r]) ? 1.0f : 0.0f;   // E[r][c]
        dg += e; ax += e*xo[2*r]; ay += e*xo[2*r+1];
    }
    float inv = 1.0f / fmaxf(dg, 1.0f);
    ax *= inv; ay *= inv;
    out[c] = ax*w3l[0] + ay*w3l[1] + b3[0] + xo[2*c]*w3r[0] + xo[2*c+1]*w3r[1];
    out[386 + 2*c]     = xo[2*c];   // x_out
    out[386 + 2*c + 1] = xo[2*c+1];
}

extern "C" void kernel_launch(void* const* d_in, const int* in_sizes, int n_in,
                              void* d_out, int out_size, void* d_ws, size_t ws_size,
                              hipStream_t stream) {
    (void)in_sizes; (void)n_in; (void)out_size; (void)ws_size;
    const float* nodes = (const float*)d_in[0];
    const float* A     = (const float*)d_in[1];
    const float* w1l   = (const float*)d_in[2];
    const float* b1    = (const float*)d_in[3];
    const float* w1r   = (const float*)d_in[4];
    const float* w2l   = (const float*)d_in[5];
    const float* b2    = (const float*)d_in[6];
    const float* w2r   = (const float*)d_in[7];
    const float* w3l   = (const float*)d_in[8];
    const float* b3    = (const float*)d_in[9];
    const float* w3r   = (const float*)d_in[10];
    float* out = (float*)d_out;

    char* ws = (char*)d_ws;
    float*          s_f32 = (float*)(ws);                               // 4 MB
    unsigned short* sA    = (unsigned short*)(ws + (4u << 20));         // 2 MB
    unsigned short* sB    = (unsigned short*)(ws + (6u << 20));         // 2 MB
    unsigned short* MB    = (unsigned short*)(ws + (8u << 20));         // 2 MB
    char* sm = ws + (10u << 20);
    float* deg     = (float*)(sm);                          // 32 KB
    float* aggT    = (float*)(sm + 32768);                  // 64 KB
    float* x       = (float*)(sm + 32768 + 65536);          // 64 KB
    float* G       = (float*)(sm + 32768 + 2*65536);        // 64 KB
    float* P       = (float*)(sm + 32768 + 3*65536);        // 64 KB
    float* xo      = (float*)(sm + 32768 + 4*65536);        // 1 KB (256 f)
    float* parts   = (float*)(sm + 32768 + 4*65536 + 1024);
    float* colPart = parts;            // 64
    float* tAsPart = parts + 64;       // 512
    float* entPart = parts + 576;      // 1024
    float* xop     = parts + 1600;     // 4096
    char* sm2 = sm + 327680;                                // 320 KB into sm
    float* degP    = (float*)(sm2);                         // 128 KB (4 x 8192)
    float* aggP    = (float*)(sm2 + 131072);                // 256 KB (4 x 16384)
    float* Gp      = (float*)(sm2 + 131072 + 262144);       // 512 KB (8 x 16384)
    float* Pp      = (float*)(sm2 + 131072 + 262144 + 524288);  // 512 KB

    k_colreduce<<<dim3(16,4), 256, 0, stream>>>(A, nodes, degP, aggP, colPart);
    k_degred   <<<32,   256, 0, stream>>>(degP, aggP, deg, aggT);
    k_softmax  <<<1024, 256, 0, stream>>>(nodes, deg, aggT, w1l, b1, w1r,
                                          w2l, b2, w2r, s_f32, sA, sB, x, entPart);
    k_AxS      <<<512,  256, 0, stream>>>(A, sB, s_f32, MB, tAsPart);
    k_sTB      <<<dim3(8,8), 256, 0, stream>>>(sA, sB, Gp);
    k_sTB      <<<dim3(8,8), 256, 0, stream>>>(sA, MB, Pp);
    k_red8     <<<64,   256, 0, stream>>>(Gp, G);
    k_red8     <<<64,   256, 0, stream>>>(Pp, P);
    k_xout     <<<16,   256, 0, stream>>>(s_f32, x, xop);
    k_xored    <<<1,    256, 0, stream>>>(xop, xo);
    k_scalars  <<<1,    256, 0, stream>>>(G, colPart, tAsPart, entPart, out);
    k_final    <<<1,    128, 0, stream>>>(P, xo, w3l, b3, w3r, out);
}
